// SentimentNetwork_55834574848024
// MI455X (gfx1250) — matrix operand order repaired and checked by reading, not compiled, not run on hardware
//
#include <hip/hip_runtime.h>
#include <hip/hip_bf16.h>
#include <string.h>

// Problem constants
#define VV   25000
#define EE   300
#define EP   320      // E padded to multiple of 32 (zero-filled)
#define HH   512
#define G4   2048     // 4*H
#define BB   128
#define TT   512

#define NKC_H (HH / 32)        // 16 K-chunks for recurrent GEMM
#define NKC_E (EP / 32)        // 10 K-chunks for embedding GEMM
#define LDS_WHH_BYTES (4 * NKC_H * 32 * 32)   // 65536
#define LDS_WIH_BYTES (4 * NKC_E * 32 * 32)   // 40960
#define LDS_TOTAL     (LDS_WHH_BYTES + LDS_WIH_BYTES)  // 106496 < 320KB WGP LDS

typedef __attribute__((ext_vector_type(16))) __bf16 v16bf;
typedef __attribute__((ext_vector_type(8)))  float  v8f;

union Frag32B { uint4 u4[2]; v16bf v; };

// ---------------------------------------------------------------------------
// A-matrix fragment (16x32 bf16, M x K). Lane holds row M = lane&15.
// K per element e: K = kbase + 8*(lane>=16) + (e<8 ? e : e+8)
// => two contiguous 16B global loads.
// ---------------------------------------------------------------------------
__device__ inline v16bf load_frag_A(const __bf16* __restrict__ rowptr, int kbase, int lane) {
    int off = (lane >> 4) << 3;          // 0 or 8
    const __bf16* p = rowptr + kbase + off;
    Frag32B f;
    f.u4[0] = *reinterpret_cast<const uint4*>(p);
    f.u4[1] = *reinterpret_cast<const uint4*>(p + 16);
    return f.v;
}

// ---------------------------------------------------------------------------
// B-matrix fragment from LDS, stored in fragment order: for each (q,kc),
// lane L's 32 bytes live at ((q*nkc + kc)*32 + L)*32. Consecutive lanes read
// consecutive 32B => ds_load_b128 pairs at full LDS bandwidth.
// ---------------------------------------------------------------------------
__device__ inline v16bf lds_frag_B(const char* base, int q, int nkc, int kc, int lane) {
    const uint4* p = reinterpret_cast<const uint4*>(
        base + ((size_t)((q * nkc + kc) * 32 + lane) << 5));
    Frag32B f;
    f.u4[0] = p[0];
    f.u4[1] = p[1];
    return f.v;
}

__device__ inline float sigm(float x) { return 1.0f / (1.0f + __expf(-x)); }
__device__ inline float tanh_fast(float x) {
    float e2 = __expf(2.0f * x);
    return (e2 - 1.0f) / (e2 + 1.0f);
}

// ---------------------------------------------------------------------------
// Prep: f32 -> bf16 with column padding (zero-fill beyond `cols`).
// ---------------------------------------------------------------------------
__global__ void cvt_pad_kernel(const float* __restrict__ src, __bf16* __restrict__ dst,
                               int rows, int cols, int colsp) {
    int i = blockIdx.x * blockDim.x + threadIdx.x;
    if (i >= rows * colsp) return;
    int r = i / colsp, c = i - r * colsp;
    dst[i] = (c < cols) ? (__bf16)src[r * cols + c] : (__bf16)0.0f;
}

__global__ void prep_bias_kernel(const float* __restrict__ bih, const float* __restrict__ bhh,
                                 float* __restrict__ bias, int n) {
    int i = blockIdx.x * blockDim.x + threadIdx.x;
    if (i < n) bias[i] = bih[i] + bhh[i];
}

__global__ void init_state_kernel(const float* __restrict__ h0, const float* __restrict__ c0,
                                  __bf16* __restrict__ hA, unsigned int* __restrict__ bar,
                                  int n) {
    int i = blockIdx.x * blockDim.x + threadIdx.x;
    if (i < n) hA[i] = (__bf16)h0[i];
    if (i < TT) bar[i] = 0u;
    (void)c0;
}

// ---------------------------------------------------------------------------
// Persistent LSTM: one launch runs all T=512 steps.
// Grid: 32 blocks (one h-column group of 16 each, pinned to one WGP),
// 256 threads = 8 waves. Wave w owns b-rows [16w,16w+16) x its 16 h-cols for
// the WHOLE sequence => cell state c stays in 8 VGPRs per lane.
// Weights are staged into LDS ONCE in fragment order and re-read via DS for
// all 512 steps. Steps are separated by a device-scope release/acquire
// barrier on bar[t] (fresh counter per step => no reset races).
// ---------------------------------------------------------------------------
__global__ __launch_bounds__(256)
void lstm_persistent_kernel(const int*    __restrict__ x,
                            const __bf16* __restrict__ embB,
                            const __bf16* __restrict__ WihB,
                            const __bf16* __restrict__ WhhB,
                            const float*  __restrict__ bias,
                            const float*  __restrict__ c0,
                            __bf16*       __restrict__ hA,
                            __bf16*       __restrict__ hB,
                            unsigned int* __restrict__ bar,
                            float*        __restrict__ h_outf,   // d_out + 128
                            float*        __restrict__ c_outf) { // d_out + 128 + B*H
    extern __shared__ __align__(16) char s_lds[];

    const int lane  = threadIdx.x & 31;
    const int wave  = threadIdx.x >> 5;
    const int cg    = blockIdx.x;            // 0..31 : h-column group
    const int mrow0 = wave << 4;             // first b-row of this wave's tile
    const int nloc  = lane & 15;
    const int hcol  = (cg << 4) + nloc;      // global h column (0..511)
    const int mhalf = (lane >> 4) << 3;      // C/D layout: M = r + 8*(lane>=16)

    // ---- One-time LDS staging of this block's weight slice (frag order) ----
    // Whh: 4q * 16n * 64 k16-chunks (16B each) = 4096 chunks
    for (int c = threadIdx.x; c < 4096; c += 256) {
        int k16 = c & 63;
        int n   = (c >> 6) & 15;
        int q   = c >> 10;
        int g   = q * HH + (cg << 4) + n;
        uint4 d = *reinterpret_cast<const uint4*>(WhhB + (size_t)g * HH + k16 * 8);
        int kc = k16 >> 2, r = k16 & 3;
        int fl = (r & 1) ? (n + 16) : n;     // lane slot (koff half)
        int sl = r >> 1;                     // 16B slot within lane's 32B
        *reinterpret_cast<uint4*>(s_lds + (((q * NKC_H + kc) * 32 + fl) << 5) + (sl << 4)) = d;
    }
    // Wih: 4q * 16n * 40 k16-chunks = 2560 chunks
    for (int c = threadIdx.x; c < 2560; c += 256) {
        int k16 = c % 40;
        int n   = (c / 40) & 15;
        int q   = c / 640;
        int g   = q * HH + (cg << 4) + n;
        uint4 d = *reinterpret_cast<const uint4*>(WihB + (size_t)g * EP + k16 * 8);
        int kc = k16 >> 2, r = k16 & 3;
        int fl = (r & 1) ? (n + 16) : n;
        int sl = r >> 1;
        *reinterpret_cast<uint4*>(s_lds + LDS_WHH_BYTES +
                                  (((q * NKC_E + kc) * 32 + fl) << 5) + (sl << 4)) = d;
    }
    __syncthreads();

    // ---- Per-lane constants: biases and resident cell state ----
    const float bI = bias[0 * HH + hcol];
    const float bF = bias[1 * HH + hcol];
    const float bG = bias[2 * HH + hcol];
    const float bO = bias[3 * HH + hcol];

    v8f c_reg;
#pragma unroll
    for (int r = 0; r < 8; ++r)
        c_reg[r] = c0[(size_t)(mrow0 + r + mhalf) * HH + hcol];

    const int arow_b = mrow0 + nloc;         // A-matrix row (M = lane&15)

    for (int t = 0; t < TT; ++t) {
        const __bf16* h_cur = (t & 1) ? hB : hA;
        __bf16*       h_nxt = (t & 1) ? hA : hB;

        v8f acc[4] = {};

        // Phase 1: embedding GEMM (K over EP=320, zero-padded)
        const int token = x[arow_b * TT + t];
        const __bf16* arow = embB + (size_t)token * EP;
#pragma unroll
        for (int kc = 0; kc < NKC_E; ++kc) {
            v16bf a = load_frag_A(arow, kc * 32, lane);
#pragma unroll
            for (int q = 0; q < 4; ++q) {
                v16bf b = lds_frag_B(s_lds + LDS_WHH_BYTES, q, NKC_E, kc, lane);
                acc[q] = __builtin_amdgcn_wmma_f32_16x16x32_bf16(
                             false, a, false, b, (short)0, acc[q], false, false);
            }
        }

        // Phase 2: recurrent GEMM (K over H=512)
        const __bf16* hrow = h_cur + (size_t)arow_b * HH;
#pragma unroll 4
        for (int kc = 0; kc < NKC_H; ++kc) {
            v16bf a = load_frag_A(hrow, kc * 32, lane);
#pragma unroll
            for (int q = 0; q < 4; ++q) {
                v16bf b = lds_frag_B(s_lds, q, NKC_H, kc, lane);
                acc[q] = __builtin_amdgcn_wmma_f32_16x16x32_bf16(
                             false, a, false, b, (short)0, acc[q], false, false);
            }
        }

        // Elementwise LSTM cell update (c in registers)
        const int last = (t == TT - 1);
#pragma unroll
        for (int r = 0; r < 8; ++r) {
            const size_t idx = (size_t)(mrow0 + r + mhalf) * HH + hcol;
            float ig = sigm(acc[0][r] + bI);
            float fg = sigm(acc[1][r] + bF);
            float gg = tanh_fast(acc[2][r] + bG);
            float og = sigm(acc[3][r] + bO);
            float cn = fg * c_reg[r] + ig * gg;
            c_reg[r] = cn;
            float hn = og * tanh_fast(cn);
            h_nxt[idx] = (__bf16)hn;
            if (last) { h_outf[idx] = hn; c_outf[idx] = cn; }
        }

        // ---- Device-scope step barrier (fresh counter per step) ----
        __threadfence();                     // flush this thread's h stores
        __syncthreads();                     // all threads in block fenced
        if (threadIdx.x == 0) {
            __hip_atomic_fetch_add(&bar[t], 1u, __ATOMIC_RELEASE,
                                   __HIP_MEMORY_SCOPE_AGENT);
            while (__hip_atomic_load(&bar[t], __ATOMIC_ACQUIRE,
                                     __HIP_MEMORY_SCOPE_AGENT) < 32u) {
                __builtin_amdgcn_s_sleep(2);
            }
        }
        __syncthreads();
        __builtin_amdgcn_fence(__ATOMIC_ACQUIRE, "agent");  // invalidate stale lines
    }
}

// ---------------------------------------------------------------------------
// Finalize: y[b] = h_T[b,:] . fc_w + fc_b (h_T and c_T already in d_out).
// ---------------------------------------------------------------------------
__global__ void finalize_kernel(const float* __restrict__ h_out,   // = d_out+128
                                const float* __restrict__ fc_w,
                                const float* __restrict__ fc_b,
                                float* __restrict__ d_out) {
    int tid = blockIdx.x * blockDim.x + threadIdx.x;
    if (tid < BB) {
        float s = fc_b[0];
        const float* hr = h_out + tid * HH;
        for (int k = 0; k < HH; ++k) s += hr[k] * fc_w[k];
        d_out[tid] = s;
    }
}

// ---------------------------------------------------------------------------
// Workspace layout (bytes)
// ---------------------------------------------------------------------------
#define OFF_EMB   ((size_t)0)                                  // V*EP bf16 = 16,000,000
#define OFF_WIH   (OFF_EMB + (size_t)VV * EP * 2)              // G4*EP bf16
#define OFF_WHH   (OFF_WIH + (size_t)G4 * EP * 2)              // G4*H  bf16
#define OFF_BIAS  (OFF_WHH + (size_t)G4 * HH * 2)              // G4 f32
#define OFF_HA    (OFF_BIAS + (size_t)G4 * 4)                  // B*H bf16
#define OFF_HB    (OFF_HA + (size_t)BB * HH * 2)               // B*H bf16
#define OFF_BAR   (OFF_HB + (size_t)BB * HH * 2)               // TT u32

extern "C" void kernel_launch(void* const* d_in, const int* in_sizes, int n_in,
                              void* d_out, int out_size, void* d_ws, size_t ws_size,
                              hipStream_t stream) {
    const int*   x    = (const int*)  d_in[0];
    const float* h0   = (const float*)d_in[1];
    const float* c0   = (const float*)d_in[2];
    const float* emb  = (const float*)d_in[3];
    const float* Wih  = (const float*)d_in[4];
    const float* Whh  = (const float*)d_in[5];
    const float* bih  = (const float*)d_in[6];
    const float* bhh  = (const float*)d_in[7];
    const float* fcw  = (const float*)d_in[8];
    const float* fcb  = (const float*)d_in[9];
    float* out = (float*)d_out;

    char* ws = (char*)d_ws;
    __bf16* embB = (__bf16*)(ws + OFF_EMB);
    __bf16* WihB = (__bf16*)(ws + OFF_WIH);
    __bf16* WhhB = (__bf16*)(ws + OFF_WHH);
    float*  bias = (float*) (ws + OFF_BIAS);
    __bf16* hA   = (__bf16*)(ws + OFF_HA);
    __bf16* hB   = (__bf16*)(ws + OFF_HB);
    unsigned int* bar = (unsigned int*)(ws + OFF_BAR);

    // Allow >default dynamic LDS for the persistent kernel (idempotent).
    (void)hipFuncSetAttribute((const void*)lstm_persistent_kernel,
                              hipFuncAttributeMaxDynamicSharedMemorySize, LDS_TOTAL);

    // Weight / embedding conversion to bf16 (padded K for the embedding GEMM)
    {
        int n = VV * EP;
        cvt_pad_kernel<<<(n + 255) / 256, 256, 0, stream>>>(emb, embB, VV, EE, EP);
    }
    {
        int n = G4 * EP;
        cvt_pad_kernel<<<(n + 255) / 256, 256, 0, stream>>>(Wih, WihB, G4, EE, EP);
    }
    {
        int n = G4 * HH;
        cvt_pad_kernel<<<(n + 255) / 256, 256, 0, stream>>>(Whh, WhhB, G4, HH, HH);
    }
    prep_bias_kernel<<<(G4 + 255) / 256, 256, 0, stream>>>(bih, bhh, bias, G4);
    init_state_kernel<<<(BB * HH + 255) / 256, 256, 0, stream>>>(h0, c0, hA, bar, BB * HH);

    // One persistent launch runs the entire recurrence.
    lstm_persistent_kernel<<<32, 256, LDS_TOTAL, stream>>>(
        x, embB, WihB, WhhB, bias, c0, hA, hB, bar,
        out + 128, out + 128 + BB * HH);

    finalize_kernel<<<1, 256, 0, stream>>>(out + 128, fcw, fcb, out);
}
